// SubstrateConvLayer_37941741093155
// MI455X (gfx1250) — compile-verified, matrix operand order checked by reading, not run on hardware
//
#include <hip/hip_runtime.h>
#include <hip/hip_bf16.h>
#include <math.h>

// CDNA5 / gfx1250: wave32, WMMA 16x16x32 bf16 with f32 accumulate,
// async direct-to-LDS gather staging (ASYNCcnt).
typedef __attribute__((ext_vector_type(16))) __bf16 v16bf;
typedef __attribute__((ext_vector_type(8)))  float  v8f;

#define ASTR    200      // padded LDS row stride (in bf16 elems) for A slab and W slab
#define O_DIM   64
#define IN_DIM  192
#define OUT_DIM 128
#define EPSI    1e-5f

#if defined(__has_builtin)
#  if __has_builtin(__builtin_amdgcn_global_load_async_to_lds_b128)
#    define HAVE_ASYNC_LDS 1
#  endif
#endif
#ifndef HAVE_ASYNC_LDS
#  define HAVE_ASYNC_LDS 0
#endif

// exact pointee type expected by the async-LDS builtin: int vector_size(16)
typedef int v4i __attribute__((vector_size(16)));
typedef __attribute__((address_space(1))) v4i as1_v4i;   // prints as "__device__"
typedef __attribute__((address_space(3))) v4i as3_v4i;   // LDS

// mem -> LDS, per-lane, no VGPR round trip (GLOBAL_LOAD_ASYNC_TO_LDS_B128)
__device__ __forceinline__ void async_copy_b128(const void* g, void* l) {
#if HAVE_ASYNC_LDS
  __builtin_amdgcn_global_load_async_to_lds_b128((as1_v4i*)g, (as3_v4i*)l, 0, 0);
#else
  *(uint4*)l = *(const uint4*)g;
#endif
}
__device__ __forceinline__ void wait_async_lds() {
#if HAVE_ASYNC_LDS
  asm volatile("s_wait_asynccnt 0x0" ::: "memory");   // ASYNCcnt == 0
#endif
}

__device__ __forceinline__ unsigned short f2b(float f) {
  unsigned u = __float_as_uint(f);
  u = (u + 0x7FFFu + ((u >> 16) & 1u)) >> 16;   // round-to-nearest-even bf16
  return (unsigned short)u;
}
__device__ __forceinline__ float softplusf(float x) {
  return fmaxf(x, 0.f) + log1pf(__expf(-fabsf(x)));
}
__device__ __forceinline__ float sigmoidf(float x) {
  return 1.f / (1.f + __expf(-x));
}

// ---- pass 0: fp32 -> bf16 conversion of W, n_feat, e_feat ------------------
__global__ void convert_kernel(const float* __restrict__ W,
                               const float* __restrict__ nf,
                               const float* __restrict__ ef,
                               unsigned short* __restrict__ wbf,
                               unsigned short* __restrict__ nbf,
                               unsigned short* __restrict__ ebf,
                               long wN, long nN, long eN) {
  long stride = (long)gridDim.x * blockDim.x;
  for (long i = (long)blockIdx.x * blockDim.x + threadIdx.x; i < eN; i += stride) {
    ebf[i] = f2b(ef[i]);
    if (i < nN) nbf[i] = f2b(nf[i]);
    if (i < wN) wbf[i] = f2b(W[i]);
  }
}

// ---- edge GEMM: 16 edges/wave, h[16x128] = cat[16x192] x W^T[192x128] ------
// PASS 1: accumulate per-channel sum / sumsq of h (for BN-over-E stats)
// PASS 2: recompute h, BN + sigmoid*softplus gate, scatter-add msg into agg
template <int PASS>
__global__ __launch_bounds__(256) void edge_gemm_kernel(
    const unsigned short* __restrict__ wbf,
    const unsigned short* __restrict__ nbf,
    const unsigned short* __restrict__ ebf,
    const int* __restrict__ srcI, const int* __restrict__ dstI,
    const float* __restrict__ bias, int E,
    float* __restrict__ gsum, float* __restrict__ gsq,
    const float* __restrict__ mean1, const float* __restrict__ rstd1,
    const float* __restrict__ g1, const float* __restrict__ beta1,
    float* __restrict__ agg) {
  __shared__ unsigned short wsh[OUT_DIM * ASTR];   // 51200 B, bf16 W, rows padded
  __shared__ unsigned short ash[8][16 * ASTR];     // 51200 B, per-wave A slabs
  __shared__ float ssum[OUT_DIM], ssq[OUT_DIM];

  // async direct-to-LDS fill of W (bf16, 48KB useful; L2-resident source)
  for (int i = threadIdx.x; i < OUT_DIM * (IN_DIM / 8); i += blockDim.x) {
    int o = i / (IN_DIM / 8), p = i % (IN_DIM / 8);
    async_copy_b128(&wbf[o * IN_DIM + p * 8], &wsh[o * ASTR + p * 8]);
  }
  if (PASS == 1 && threadIdx.x < OUT_DIM) { ssum[threadIdx.x] = 0.f; ssq[threadIdx.x] = 0.f; }
  wait_async_lds();
  __syncthreads();

  const int wave = threadIdx.x >> 5;
  const int lane = threadIdx.x & 31;
  const int half = lane >> 4;      // K-half select per ISA A/B bf16 layout
  const int nm   = lane & 15;      // A: row m; B/C/D: col n
  const int e0   = (blockIdx.x * 8 + wave) * 16;

  unsigned short* a = ash[wave];

  // gather 16 edge rows [dst | src | edge] as bf16 straight into LDS
  for (int i = lane; i < 16 * 8; i += 32) {
    int r = i >> 3, p = i & 7;
    int e = e0 + r;
    if (e < E) {
      long d = dstI[e], s = srcI[e];
      async_copy_b128(&nbf[d * O_DIM + p * 8],        &a[r * ASTR +       p * 8]);
      async_copy_b128(&nbf[s * O_DIM + p * 8],        &a[r * ASTR +  64 + p * 8]);
      async_copy_b128(&ebf[(long)e * O_DIM + p * 8],  &a[r * ASTR + 128 + p * 8]);
    } else {
      uint4 z = {0, 0, 0, 0};
      *(uint4*)&a[r * ASTR +       p * 8] = z;
      *(uint4*)&a[r * ASTR +  64 + p * 8] = z;
      *(uint4*)&a[r * ASTR + 128 + p * 8] = z;
    }
  }
  wait_async_lds();

  union Frag { v16bf v; unsigned u[8]; };
  v8f acc[8];
  v8f zero = {0.f, 0.f, 0.f, 0.f, 0.f, 0.f, 0.f, 0.f};
#pragma unroll
  for (int t = 0; t < 8; ++t) acc[t] = zero;

  const unsigned* arow = (const unsigned*)&a[nm * ASTR];   // nm*400B, dword aligned
#pragma unroll
  for (int kk = 0; kk < 6; ++kk) {
    const int k0 = kk * 32 + half * 8;   // even element index -> dword k0/2
    Frag av;
#pragma unroll
    for (int i = 0; i < 4; ++i) {
      av.u[i]     = arow[(k0 >> 1) + i];        // K = k0 .. k0+7
      av.u[4 + i] = arow[(k0 >> 1) + 8 + i];    // K = k0+16 .. k0+23
    }
#pragma unroll
    for (int t = 0; t < 8; ++t) {
      const unsigned* wrow = (const unsigned*)&wsh[(t * 16 + nm) * ASTR];
      Frag bv;
#pragma unroll
      for (int i = 0; i < 4; ++i) {
        bv.u[i]     = wrow[(k0 >> 1) + i];
        bv.u[4 + i] = wrow[(k0 >> 1) + 8 + i];
      }
      acc[t] = __builtin_amdgcn_wmma_f32_16x16x32_bf16(
          false, av.v, false, bv.v, (short)0, acc[t], false, false);
    }
  }

  if (PASS == 1) {
#pragma unroll
    for (int t = 0; t < 8; ++t) {
      int o = t * 16 + nm;
      float bo = bias[o];
      float s = 0.f, q = 0.f;
#pragma unroll
      for (int r = 0; r < 8; ++r) {            // reg r holds edge m = r + 8*half
        int e = e0 + r + half * 8;
        if (e < E) { float v = acc[t][r] + bo; s += v; q += v * v; }
      }
      atomicAdd(&ssum[o], s);
      atomicAdd(&ssq[o], q);
    }
    __syncthreads();
    if (threadIdx.x < OUT_DIM) {
      atomicAdd(&gsum[threadIdx.x], ssum[threadIdx.x]);
      atomicAdd(&gsq[threadIdx.x],  ssq[threadIdx.x]);
    }
  } else {
    int di[8];
#pragma unroll
    for (int r = 0; r < 8; ++r) {
      int e = e0 + r + half * 8;
      di[r] = (e < E) ? dstI[e] : -1;
    }
    // filt col o (tiles 0..3) and core col o+64 (tiles 4..7) live in the SAME
    // lane & reg -> gated activation is lane-local.
#pragma unroll
    for (int t = 0; t < 4; ++t) {
      int of = t * 16 + nm, oc = of + 64;
      float af = rstd1[of] * g1[of];
      float cf = (bias[of] - mean1[of]) * af + beta1[of];
      float ac = rstd1[oc] * g1[oc];
      float cc = (bias[oc] - mean1[oc]) * ac + beta1[oc];
#pragma unroll
      for (int r = 0; r < 8; ++r) {
        if (di[r] >= 0) {
          float f = acc[t][r]     * af + cf;
          float c = acc[t + 4][r] * ac + cc;
          float m = sigmoidf(f) * softplusf(c);
          atomicAdd(&agg[(long)di[r] * O_DIM + of], m);   // global_atomic_add_f32
        }
      }
    }
  }
}

// ---- BN stat finalization: mean / rstd from sum / sumsq --------------------
__global__ void finalize_stats_kernel(const float* __restrict__ sum,
                                      const float* __restrict__ sq,
                                      float* __restrict__ mean,
                                      float* __restrict__ rstd,
                                      float invCount, int C) {
  int i = threadIdx.x;
  if (i < C) {
    float m = sum[i] * invCount;
    float v = sq[i] * invCount - m * m;
    mean[i] = m;
    rstd[i] = rsqrtf(fmaxf(v, 0.f) + EPSI);
  }
}

// ---- per-channel sum/sumsq over agg[N,64] ----------------------------------
__global__ void agg_stats_kernel(const float* __restrict__ agg,
                                 float* __restrict__ gsum, float* __restrict__ gsq,
                                 long total) {
  __shared__ float ss[O_DIM], sq[O_DIM];
  if (threadIdx.x < O_DIM) { ss[threadIdx.x] = 0.f; sq[threadIdx.x] = 0.f; }
  __syncthreads();
  long stride = (long)gridDim.x * blockDim.x;   // multiple of 64 -> channel fixed
  int ch = threadIdx.x & (O_DIM - 1);
  float s = 0.f, q = 0.f;
  for (long i = (long)blockIdx.x * blockDim.x + threadIdx.x; i < total; i += stride) {
    float v = agg[i]; s += v; q += v * v;
  }
  atomicAdd(&ss[ch], s);
  atomicAdd(&sq[ch], q);
  __syncthreads();
  if (threadIdx.x < O_DIM) {
    atomicAdd(&gsum[threadIdx.x], ss[threadIdx.x]);
    atomicAdd(&gsq[threadIdx.x],  sq[threadIdx.x]);
  }
}

// ---- out = softplus(BN2(agg) + n_feat) -------------------------------------
__global__ void output_kernel(const float* __restrict__ agg,
                              const float* __restrict__ nf,
                              const float* __restrict__ mean2,
                              const float* __restrict__ rstd2,
                              const float* __restrict__ g2,
                              const float* __restrict__ beta2,
                              float* __restrict__ out, long total) {
  long i = (long)blockIdx.x * blockDim.x + threadIdx.x;
  if (i < total) {
    int ch = (int)(i & (O_DIM - 1));
    float v = (agg[i] - mean2[ch]) * rstd2[ch] * g2[ch] + beta2[ch] + nf[i];
    out[i] = softplusf(v);
  }
}

extern "C" void kernel_launch(void* const* d_in, const int* in_sizes, int n_in,
                              void* d_out, int out_size, void* d_ws, size_t ws_size,
                              hipStream_t stream) {
  const float* n_feat = (const float*)d_in[0];
  const float* e_feat = (const float*)d_in[1];
  const int*   srcI   = (const int*)d_in[2];
  const int*   dstI   = (const int*)d_in[3];
  const float* W      = (const float*)d_in[4];
  const float* b      = (const float*)d_in[5];
  const float* g1     = (const float*)d_in[6];
  const float* beta1  = (const float*)d_in[7];
  const float* g2     = (const float*)d_in[8];
  const float* beta2  = (const float*)d_in[9];
  float* out = (float*)d_out;

  const long N = in_sizes[0] / O_DIM;
  const int  E = in_sizes[2];

  char* ws = (char*)d_ws;
  size_t off = 0;
  auto take = [&](size_t bytes) -> char* {
    char* p = ws + off;
    off += (bytes + 255) & ~(size_t)255;
    return p;
  };
  float* agg   = (float*)take(N * O_DIM * sizeof(float));
  float* sum1  = (float*)take(OUT_DIM * sizeof(float));
  float* sq1   = (float*)take(OUT_DIM * sizeof(float));
  float* mean1 = (float*)take(OUT_DIM * sizeof(float));
  float* rstd1 = (float*)take(OUT_DIM * sizeof(float));
  float* sum2  = (float*)take(O_DIM * sizeof(float));
  float* sq2   = (float*)take(O_DIM * sizeof(float));
  float* mean2 = (float*)take(O_DIM * sizeof(float));
  float* rstd2 = (float*)take(O_DIM * sizeof(float));
  unsigned short* wbf = (unsigned short*)take((size_t)OUT_DIM * IN_DIM * sizeof(unsigned short));
  unsigned short* nbf = (unsigned short*)take((size_t)N * O_DIM * sizeof(unsigned short));
  unsigned short* ebf = (unsigned short*)take((size_t)E * O_DIM * sizeof(unsigned short));
  (void)ws_size; (void)n_in; (void)out_size;

  (void)hipMemsetAsync(agg,  0, N * O_DIM * sizeof(float), stream);
  (void)hipMemsetAsync(sum1, 0, OUT_DIM * sizeof(float), stream);
  (void)hipMemsetAsync(sq1,  0, OUT_DIM * sizeof(float), stream);
  (void)hipMemsetAsync(sum2, 0, O_DIM * sizeof(float), stream);
  (void)hipMemsetAsync(sq2,  0, O_DIM * sizeof(float), stream);

  convert_kernel<<<2048, 256, 0, stream>>>(W, n_feat, e_feat, wbf, nbf, ebf,
                                           (long)OUT_DIM * IN_DIM, N * O_DIM, (long)E * O_DIM);

  long tiles   = ((long)E + 15) / 16;
  long gblocks = (tiles + 7) / 8;
  edge_gemm_kernel<1><<<dim3((unsigned)gblocks), 256, 0, stream>>>(
      wbf, nbf, ebf, srcI, dstI, b, E, sum1, sq1,
      nullptr, nullptr, nullptr, nullptr, nullptr);
  finalize_stats_kernel<<<1, 128, 0, stream>>>(sum1, sq1, mean1, rstd1, 1.f / (float)E, OUT_DIM);
  edge_gemm_kernel<2><<<dim3((unsigned)gblocks), 256, 0, stream>>>(
      wbf, nbf, ebf, srcI, dstI, b, E, nullptr, nullptr,
      mean1, rstd1, g1, beta1, agg);
  agg_stats_kernel<<<512, 256, 0, stream>>>(agg, sum2, sq2, N * O_DIM);
  finalize_stats_kernel<<<1, 64, 0, stream>>>(sum2, sq2, mean2, rstd2, 1.f / (float)N, O_DIM);
  long total = N * O_DIM;
  output_kernel<<<(unsigned)((total + 255) / 256), 256, 0, stream>>>(
      agg, n_feat, mean2, rstd2, g2, beta2, out, total);
}
